// MemoryModule_32272384262349
// MI455X (gfx1250) — compile-verified, matrix operand order checked by reading, not run on hardware
//
#include <hip/hip_runtime.h>
#include <hip/hip_bf16.h>
#include <stdint.h>

// ---------------------------------------------------------------------------
// Cosine-similarity nearest-memory lookup on MI455X (gfx1250):
//   sim = normalize(x) @ normalize(memory)^T ; idx = argmax_m ; out = x + memory[idx]
// B=8192, D=512, M=32768.  275 GFLOP matmul -> bf16 WMMA 16x16x32, f32 accum.
// memory fits in 192MB L2, so the GEMM is compute-bound; pipeline LDS staging
// (async LDS-DMA, ASYNCcnt) and B-fragment register loads against the WMMA chain.
// ---------------------------------------------------------------------------

typedef __attribute__((ext_vector_type(16))) __bf16 v16bf;
typedef __attribute__((ext_vector_type(8)))  float  v8f;
typedef __attribute__((ext_vector_type(4)))  int    i32x4;

#define D_DIM   512
#define TILE_B  128                 // x-rows per block = 8 waves * 16
#define NS      32                  // memory rows staged per iteration (2 N-tiles)
#define ROW_U4  64                  // 16B chunks per row (512*2/16)

#if defined(__HIP_DEVICE_COMPILE__) && \
    __has_builtin(__builtin_amdgcn_global_load_async_to_lds_b128) && \
    __has_builtin(__builtin_amdgcn_s_wait_asynccnt)
#define USE_ASYNC 1
typedef __attribute__((address_space(1))) i32x4 g_i32x4;   // global
typedef __attribute__((address_space(3))) i32x4 l_i32x4;   // LDS
#else
#define USE_ASYNC 0
#endif

union FragU { v16bf v; uint4 q[2]; };

// XOR swizzle: row-major (stride 1024B) LDS layout, 16B chunk index's low 4
// bits xored with row&15 -> pad-free and conflict-optimal for b128 reads.
static __device__ __forceinline__ int swz(int row, int cb) {
    return (cb & 0x30) | ((cb ^ row) & 0x0F);
}

static __device__ __forceinline__ uint32_t pack_bf16x2(float a, float b) {
    uint32_t ua = __builtin_bit_cast(uint32_t, a);
    uint32_t ub = __builtin_bit_cast(uint32_t, b);
    ua = (ua + 0x7FFFu + ((ua >> 16) & 1u)) >> 16;   // RNE truncate to bf16
    ub = (ub + 0x7FFFu + ((ub >> 16) & 1u)) >> 16;
    return ua | (ub << 16);
}

// --- Kernel 1: per-row L2 normalize (fp32 in) -> bf16 out. One wave per row.
__global__ void k_normalize(const float* __restrict__ src,
                            uint32_t* __restrict__ dst_bf16_as_u32, int nrows) {
    int wave = (int)((blockIdx.x * blockDim.x + threadIdx.x) >> 5);
    int lane = (int)(threadIdx.x & 31);
    if (wave >= nrows) return;

    const float4* row = (const float4*)(src + (size_t)wave * D_DIM);
    float4 v[4];
    float ss = 0.f;
#pragma unroll
    for (int i = 0; i < 4; ++i) {
        v[i] = row[lane + i * 32];
        ss += v[i].x * v[i].x + v[i].y * v[i].y + v[i].z * v[i].z + v[i].w * v[i].w;
    }
#pragma unroll
    for (int off = 16; off >= 1; off >>= 1) ss += __shfl_xor(ss, off, 32);
    float inv = 1.0f / fmaxf(sqrtf(ss), 1e-8f);

    uint2* drow = (uint2*)(dst_bf16_as_u32 + (size_t)wave * (D_DIM / 2));
#pragma unroll
    for (int i = 0; i < 4; ++i) {
        uint2 p;
        p.x = pack_bf16x2(v[i].x * inv, v[i].y * inv);
        p.y = pack_bf16x2(v[i].z * inv, v[i].w * inv);
        drow[lane + i * 32] = p;
    }
}

// --- Kernel 2: zero the packed (key||~idx) argmax accumulators
__global__ void k_init_best(unsigned long long* __restrict__ best, int n) {
    int i = (int)(blockIdx.x * blockDim.x + threadIdx.x);
    if (i < n) best[i] = 0ull;
}

// --- Kernel 3: bf16 WMMA matmul with fused running argmax.
__global__ void __launch_bounds__(256)
k_simargmax(const __bf16* __restrict__ xn, const __bf16* __restrict__ mn,
            unsigned long long* __restrict__ best, int M, int mslice) {
    __shared__ uint4 ldsbuf[2][NS * ROW_U4];   // 2 x 32KB, double buffered

    const int tid  = (int)threadIdx.x;
    const int wave = tid >> 5;
    const int lane = tid & 31;
    const int half = lane >> 4;        // 0: lanes 0-15, 1: lanes 16-31
    const int l    = lane & 15;

    const int rowBase = (int)blockIdx.x * TILE_B + wave * 16;
    const int mBegin  = (int)blockIdx.y * mslice;
    const int mEnd    = mBegin + mslice;

    // ---- A tile (16 rows x 512 K) resident in VGPRs, exact ISA fragment layout:
    //  lanes 0-15: M=l, K=kbase+{0..7}(q0), kbase+{16..23}(q1); lanes 16-31: K+8.
    v16bf afrag[16];
    {
        const uint4* arow = (const uint4*)(xn + (size_t)(rowBase + l) * D_DIM);
#pragma unroll
        for (int ks = 0; ks < 16; ++ks) {
            FragU f;
            f.q[0] = arow[ks * 4 + half];
            f.q[1] = arow[ks * 4 + 2 + half];
            afrag[ks] = f.v;
        }
    }

    float    bestVal[8];
    uint32_t bestIdx[8];
#pragma unroll
    for (int i = 0; i < 8; ++i) { bestVal[i] = -3.0e38f; bestIdx[i] = 0u; }

#if USE_ASYNC
    // ---- Stage chunk via gfx1250 async LDS-DMA (ASYNCcnt), per-lane 16B moves.
#define STAGE(BUF, MB)                                                         \
    {                                                                          \
        const __bf16* mbase = mn + (size_t)(MB) * D_DIM;                       \
        _Pragma("unroll")                                                      \
        for (int k = 0; k < 8; ++k) {                                          \
            int id = tid + k * 256;                                            \
            int r = id >> 6, cb = id & 63;                                     \
            const uint4* gsrc = (const uint4*)(mbase + (size_t)r * D_DIM) + cb;\
            __builtin_amdgcn_global_load_async_to_lds_b128(                    \
                (g_i32x4*)gsrc,                                                \
                (l_i32x4*)&ldsbuf[BUF][r * ROW_U4 + swz(r, cb)], 0, 0);        \
        }                                                                      \
    }
    STAGE(0, mBegin);
    __builtin_amdgcn_s_wait_asynccnt(0);
    __syncthreads();
#else
    // ---- Fallback: VGPR round-trip, global loads for t+1 issued before compute(t).
    uint4 g[8];
#define PREFETCH(MB)                                                           \
    {                                                                          \
        const __bf16* mbase = mn + (size_t)(MB) * D_DIM;                       \
        _Pragma("unroll")                                                      \
        for (int k = 0; k < 8; ++k) {                                          \
            int id = tid + k * 256;                                            \
            int r = id >> 6, cb = id & 63;                                     \
            g[k] = ((const uint4*)(mbase + (size_t)r * D_DIM))[cb];            \
        }                                                                      \
    }
    PREFETCH(mBegin);
#endif

    int buf = 0;
    for (int mb = mBegin; mb < mEnd; mb += NS) {
#if USE_ASYNC
        if (mb + NS < mEnd) STAGE(buf ^ 1, mb + NS);   // overlaps compute below
#else
        // write previously prefetched chunk into this iteration's buffer
#pragma unroll
        for (int k = 0; k < 8; ++k) {
            int id = tid + k * 256;
            int r = id >> 6, cb = id & 63;
            ldsbuf[buf][r * ROW_U4 + swz(r, cb)] = g[k];
        }
        if (mb + NS < mEnd) PREFETCH(mb + NS);         // overlaps compute below
        __syncthreads();
#endif

        const uint4* r0 = &ldsbuf[buf][(0 + l) * ROW_U4];   // cols mb+0..15
        const uint4* r1 = &ldsbuf[buf][(16 + l) * ROW_U4];  // cols mb+16..31

        // B fragments double-buffered in registers: load ks+1 before WMMA(ks)
        FragU b0[2], b1[2];
        b0[0].q[0] = r0[swz(l, half)];     b0[0].q[1] = r0[swz(l, 2 + half)];
        b1[0].q[0] = r1[swz(l, half)];     b1[0].q[1] = r1[swz(l, 2 + half)];
        v8f c0 = {}, c1 = {};
#pragma unroll
        for (int ks = 0; ks < 16; ++ks) {
            const int cur = ks & 1;
            if (ks < 15) {
                const int c0q = (ks + 1) * 4 + half;
                b0[cur ^ 1].q[0] = r0[swz(l, c0q)];
                b0[cur ^ 1].q[1] = r0[swz(l, c0q + 2)];
                b1[cur ^ 1].q[0] = r1[swz(l, c0q)];
                b1[cur ^ 1].q[1] = r1[swz(l, c0q + 2)];
            }
            c0 = __builtin_amdgcn_wmma_f32_16x16x32_bf16(false, afrag[ks], false, b0[cur].v,
                                                         (short)0, c0, false, false);
            c1 = __builtin_amdgcn_wmma_f32_16x16x32_bf16(false, afrag[ks], false, b1[cur].v,
                                                         (short)0, c1, false, false);
        }

        // C layout: lane holds N=l, VGPR i holds M = half*8 + i.
        const uint32_t n0 = (uint32_t)(mb + l);
        const uint32_t n1 = (uint32_t)(mb + 16 + l);
#pragma unroll
        for (int i = 0; i < 8; ++i) {
            float v0 = c0[i];
            if (v0 > bestVal[i]) { bestVal[i] = v0; bestIdx[i] = n0; }
            float v1 = c1[i];
            if (v1 > bestVal[i]) { bestVal[i] = v1; bestIdx[i] = n1; }
        }

#if USE_ASYNC
        __builtin_amdgcn_s_wait_asynccnt(0);   // own stores to buf^1 complete
        __syncthreads();                       // everyone done with buf
#else
        __syncthreads();
#endif
        buf ^= 1;
    }

    // Reduce (val, idx) across the 16 lanes of each half; ties -> lowest index.
#pragma unroll
    for (int i = 0; i < 8; ++i) {
        float    v  = bestVal[i];
        uint32_t id = bestIdx[i];
#pragma unroll
        for (int off = 8; off >= 1; off >>= 1) {
            float    ov  = __shfl_xor(v, off, 16);
            uint32_t oid = (uint32_t)__shfl_xor((int)id, off, 16);
            if (ov > v || (ov == v && oid < id)) { v = ov; id = oid; }
        }
        if (l == 0) {
            int row = rowBase + half * 8 + i;
            uint32_t uv   = __builtin_bit_cast(uint32_t, v);
            uint32_t mono = (uv & 0x80000000u) ? ~uv : (uv | 0x80000000u);
            unsigned long long key =
                ((unsigned long long)mono << 32) | (unsigned long long)(~id);
            atomicMax(&best[row], key);   // cross-M-slice merge, deterministic
        }
    }
}

// --- Kernel 4: out[b] = x[b] + memory[idx[b]]  (fp32, one float4 per thread)
__global__ void k_gather(const float* __restrict__ x, const float* __restrict__ mem,
                         const unsigned long long* __restrict__ best,
                         float* __restrict__ out, int nf4) {
    int t = (int)(blockIdx.x * blockDim.x + threadIdx.x);
    if (t >= nf4) return;
    int b = t >> 7;                       // 128 float4 per row
    int c = t & 127;
    uint32_t id = ~(uint32_t)(best[b] & 0xFFFFFFFFull);
    float4 xv = ((const float4*)x)[t];
    float4 mv = ((const float4*)(mem + (size_t)id * D_DIM))[c];
    float4 o;
    o.x = xv.x + mv.x; o.y = xv.y + mv.y; o.z = xv.z + mv.z; o.w = xv.w + mv.w;
    ((float4*)out)[t] = o;
}

extern "C" void kernel_launch(void* const* d_in, const int* in_sizes, int n_in,
                              void* d_out, int out_size, void* d_ws, size_t ws_size,
                              hipStream_t stream) {
    (void)n_in; (void)out_size; (void)ws_size;
    const float* x   = (const float*)d_in[0];
    const float* mem = (const float*)d_in[1];
    const int Bn = in_sizes[0] / D_DIM;   // 8192
    const int M  = in_sizes[1] / D_DIM;   // 32768

    // Workspace layout: xn bf16 (8 MB) | mn bf16 (32 MB) | best u64 (64 KB)
    __bf16* xn = (__bf16*)d_ws;
    __bf16* mn = xn + (size_t)Bn * D_DIM;
    unsigned long long* best = (unsigned long long*)(mn + (size_t)M * D_DIM);

    k_normalize<<<(Bn + 7) / 8, 256, 0, stream>>>(x,   (uint32_t*)xn, Bn);
    k_normalize<<<(M  + 7) / 8, 256, 0, stream>>>(mem, (uint32_t*)mn, M);
    k_init_best<<<(Bn + 255) / 256, 256, 0, stream>>>(best, Bn);

    const int MSLICES = 8;                // 64 x 8 = 512 blocks -> fills WGPs
    dim3 grid(Bn / TILE_B, MSLICES);
    k_simargmax<<<grid, 256, 0, stream>>>(xn, mn, best, M, M / MSLICES);

    int nf4 = Bn * (D_DIM / 4);
    k_gather<<<(nf4 + 255) / 256, 256, 0, stream>>>(x, mem, best, (float*)d_out, nf4);
}
